// ActorCritic_44109314130284
// MI455X (gfx1250) — compile-verified, matrix operand order checked by reading, not run on hardware
//
#include <hip/hip_runtime.h>
#include <hip/hip_bf16.h>

#define DEVINL __device__ __forceinline__

typedef __attribute__((ext_vector_type(16))) __bf16 v16bf;
typedef __attribute__((ext_vector_type(8)))  float  v8f;

// Problem constants (from reference)
constexpr int Nn = 50000;
constexpr int Ee = 800000;
constexpr int Bb = 64;
constexpr int NDd = 7;
constexpr int Hh = 128;
constexpr int GFf = 10;
constexpr int NAa = 7;
constexpr float EPSc = 1e-5f;
constexpr int NTILES = Nn / 16;    // 3125 row tiles (N divisible by 16)
constexpr int CHK = 32;            // nodes per thread in chunked segment reductions

DEVINL unsigned short f2bf(float f) {
    unsigned int u = __float_as_uint(f);
    unsigned int r = u + 0x7FFFu + ((u >> 16) & 1u);   // round-to-nearest-even
    return (unsigned short)(r >> 16);
}

DEVINL void atomicMaxF(float* addr, float val) {
    unsigned int* a = (unsigned int*)addr;
    unsigned int old = *a;
    while (true) {
        float f = __uint_as_float(old);
        if (f >= val) break;
        unsigned int prev = atomicCAS(a, old, __float_as_uint(val));
        if (prev == old) break;
        old = prev;
    }
}

// ---------------- elementwise / setup kernels ----------------

__global__ void k_fill(float* p, float v, int n) {
    int i = blockIdx.x * blockDim.x + threadIdx.x;
    if (i < n) p[i] = v;
}

// h = relu(x @ in_W + in_b); also emits bf16 copy for the first WMMA GEMM
__global__ void k_input_linear(const float* __restrict__ x,
                               const float* __restrict__ W,
                               const float* __restrict__ b,
                               float* __restrict__ h,
                               unsigned short* __restrict__ hbf) {
    int i = blockIdx.x * blockDim.x + threadIdx.x;
    if (i >= Nn * Hh) return;
    int n = i >> 7, c = i & 127;
    float acc = b[c];
#pragma unroll
    for (int k = 0; k < NDd; k++) acc += x[n * NDd + k] * W[k * Hh + c];
    float r = fmaxf(acc, 0.f);
    h[i] = r;
    hbf[i] = f2bf(r);
}

__global__ void k_deg(const int* __restrict__ dst, float* __restrict__ deg) {
    int e = blockIdx.x * blockDim.x + threadIdx.x;
    if (e < Ee) atomicAdd(&deg[dst[e]], 1.f);
}

__global__ void k_dinv(const float* __restrict__ deg, float* __restrict__ dinv) {
    int n = blockIdx.x * blockDim.x + threadIdx.x;
    if (n < Nn) dinv[n] = rsqrtf(deg[n] + 1.f);
}

// Wt[c][k] = bf16(W[k][c])  (transpose + convert, 128x128)
__global__ void k_wT_bf16(const float* __restrict__ W, unsigned short* __restrict__ Wt) {
    int i = blockIdx.x * blockDim.x + threadIdx.x;
    if (i >= Hh * Hh) return;
    int c = i >> 7, k = i & 127;
    Wt[i] = f2bf(W[k * Hh + c]);
}

// ---------------- WMMA GEMM with fused self-term ----------------
// m[N x 128] = h_bf16 @ W ; agg = m * dinv^2 + conv_b (fused epilogue).
// Block = 8 waves; weight (transposed, bf16) staged in LDS; each wave owns a
// full 16x128 output slab (8 accumulator tiles) so A is loaded once.
__global__ void __launch_bounds__(256)
k_gemm_bf16(const unsigned short* __restrict__ A,
            const unsigned short* __restrict__ Bt,
            const float* __restrict__ dinv,
            const float* __restrict__ cb,
            float* __restrict__ Cm,
            float* __restrict__ agg) {
    __shared__ unsigned short sW[Hh * Hh];          // 32 KB
    // stage full transposed weight into LDS (uint4 = 8 bf16 per copy)
    {
        const uint4* src = (const uint4*)Bt;
        uint4* dst = (uint4*)sW;
        for (int i = threadIdx.x; i < (Hh * Hh) / 8; i += 256) dst[i] = src[i];
    }
    __syncthreads();

    int waveInBlk = threadIdx.x >> 5;
    int lane = threadIdx.x & 31;
    int tm = blockIdx.x * 8 + waveInBlk;            // row tile
    if (tm >= NTILES) return;                       // wave-uniform
    int half = lane >> 4;                           // 0: lanes 0-15, 1: lanes 16-31
    int l15 = lane & 15;
    const unsigned short* arow = A + (long long)(tm * 16 + l15) * Hh;

    v8f acc[8] = {};
    union U { v16bf v; unsigned int u[8]; };
#pragma unroll
    for (int kk = 0; kk < Hh; kk += 32) {
        U a;
#pragma unroll
        for (int j = 0; j < 8; j++) {
            int k0 = kk + ((j < 4) ? 0 : 16) + half * 8 + 2 * (j & 3);
            a.u[j] = *(const unsigned int*)(arow + k0);
        }
#pragma unroll
        for (int tn = 0; tn < 8; tn++) {
            const unsigned short* brow = sW + (tn * 16 + l15) * Hh;
            U b;
#pragma unroll
            for (int j = 0; j < 8; j++) {
                int k0 = kk + ((j < 4) ? 0 : 16) + half * 8 + 2 * (j & 3);
                b.u[j] = *(const unsigned int*)(brow + k0);
            }
            acc[tn] = __builtin_amdgcn_wmma_f32_16x16x32_bf16(
                false, a.v, false, b.v, (short)0, acc[tn], false, false);
        }
    }

    float dv[8];
#pragma unroll
    for (int i2 = 0; i2 < 8; i2++) dv[i2] = dinv[tm * 16 + i2 + half * 8];

#pragma unroll
    for (int tn = 0; tn < 8; tn++) {
        int col = tn * 16 + l15;
        float cbv = cb[col];
#pragma unroll
        for (int i2 = 0; i2 < 8; i2++) {
            int row = tm * 16 + i2 + half * 8;
            long long idx = (long long)row * Hh + col;
            float v = acc[tn][i2];
            Cm[idx] = v;
            agg[idx] = v * dv[i2] * dv[i2] + cbv;
        }
    }
}

// agg[dst] += m[src] * dinv[src]*dinv[dst] ; 32 lanes per edge, 4 channels/lane
__global__ void k_edge_scatter(const int* __restrict__ eidx, const float* __restrict__ m,
                               const float* __restrict__ dinv, float* __restrict__ agg) {
    long long t = (long long)blockIdx.x * blockDim.x + threadIdx.x;
    int e = (int)(t >> 5);
    if (e >= Ee) return;
    int lane = (int)t & 31;
    int s = eidx[e], d = eidx[Ee + e];
    float coef = dinv[s] * dinv[d];
    const float* ms = m + (long long)s * Hh;
    float* ad = agg + (long long)d * Hh;
#pragma unroll
    for (int j = 0; j < 4; j++) {
        int c = lane + 32 * j;
        atomicAdd(&ad[c], ms[c] * coef);
    }
}

// ---------------- segment reductions (chunked: ~32x fewer atomics) ----------

__global__ void k_seg_cnt(const int* __restrict__ batch, float* __restrict__ cnt) {
    int n = blockIdx.x * blockDim.x + threadIdx.x;
    if (n < Nn) atomicAdd(&cnt[batch[n]], 1.f);
}

__global__ void k_seg_sum(const float* __restrict__ v, const int* __restrict__ batch,
                          float* __restrict__ sum) {
    constexpr int NC = (Nn + CHK - 1) / CHK;
    int t = blockIdx.x * blockDim.x + threadIdx.x;
    if (t >= NC * Hh) return;
    int c = t & 127, chunk = t >> 7;
    int n0 = chunk * CHK;
    int n1 = min(n0 + CHK, Nn);
    int curb = batch[n0];
    float acc = 0.f;
    for (int n = n0; n < n1; n++) {
        int b = batch[n];
        if (b != curb) { atomicAdd(&sum[curb * Hh + c], acc); acc = 0.f; curb = b; }
        acc += v[(long long)n * Hh + c];
    }
    atomicAdd(&sum[curb * Hh + c], acc);
}

__global__ void k_mean(const float* __restrict__ sum, const float* __restrict__ cnt,
                       float* __restrict__ mean) {
    int i = blockIdx.x * blockDim.x + threadIdx.x;
    if (i >= Bb * Hh) return;
    mean[i] = sum[i] / fmaxf(cnt[i >> 7], 1.f);
}

__global__ void k_var_acc(const float* __restrict__ agg, const int* __restrict__ batch,
                          const float* __restrict__ mean, const float* __restrict__ na,
                          float* __restrict__ varsum) {
    constexpr int NC = (Nn + CHK - 1) / CHK;
    int t = blockIdx.x * blockDim.x + threadIdx.x;
    if (t >= NC * Hh) return;
    int c = t & 127, chunk = t >> 7;
    int n0 = chunk * CHK;
    int n1 = min(n0 + CHK, Nn);
    float nac = na[c];
    int curb = batch[n0];
    float mn = nac * mean[curb * Hh + c];
    float acc = 0.f;
    for (int n = n0; n < n1; n++) {
        int b = batch[n];
        if (b != curb) {
            atomicAdd(&varsum[curb * Hh + c], acc);
            acc = 0.f; curb = b; mn = nac * mean[curb * Hh + c];
        }
        float sub = agg[(long long)n * Hh + c] - mn;
        acc += sub * sub;
    }
    atomicAdd(&varsum[curb * Hh + c], acc);
}

__global__ void k_rs(const float* __restrict__ varsum, const float* __restrict__ cnt,
                     float* __restrict__ rs) {
    int i = blockIdx.x * blockDim.x + threadIdx.x;
    if (i >= Bb * Hh) return;
    rs[i] = rsqrtf(varsum[i] / fmaxf(cnt[i >> 7], 1.f) + EPSc);
}

// h = relu(nw*sub*rs + nb) + h_old ; also emits bf16 h for the next GEMM
__global__ void k_norm_finish(const float* __restrict__ agg, const int* __restrict__ batch,
                              const float* __restrict__ mean, const float* __restrict__ rs,
                              const float* __restrict__ na, const float* __restrict__ nw,
                              const float* __restrict__ nb, float* __restrict__ h,
                              unsigned short* __restrict__ hbf) {
    int i = blockIdx.x * blockDim.x + threadIdx.x;
    if (i >= Nn * Hh) return;
    int n = i >> 7, c = i & 127, b = batch[n];
    float sub = agg[i] - na[c] * mean[b * Hh + c];
    float gn = nw[c] * sub * rs[b * Hh + c] + nb[c];
    float r = fmaxf(gn, 0.f) + h[i];
    h[i] = r;
    hbf[i] = f2bf(r);
}

__global__ void k_gmax(const float* __restrict__ h, const int* __restrict__ batch,
                       float* __restrict__ gmax) {
    constexpr int NC = (Nn + CHK - 1) / CHK;
    int t = blockIdx.x * blockDim.x + threadIdx.x;
    if (t >= NC * Hh) return;
    int c = t & 127, chunk = t >> 7;
    int n0 = chunk * CHK;
    int n1 = min(n0 + CHK, Nn);
    int curb = batch[n0];
    float acc = -3.0e38f;
    for (int n = n0; n < n1; n++) {
        int b = batch[n];
        if (b != curb) { atomicMaxF(&gmax[curb * Hh + c], acc); acc = -3.0e38f; curb = b; }
        acc = fmaxf(acc, h[(long long)n * Hh + c]);
    }
    atomicMaxF(&gmax[curb * Hh + c], acc);
}

__global__ void k_emb(const float* __restrict__ mean, const float* __restrict__ gmax,
                      float* __restrict__ emb) {
    int i = blockIdx.x * blockDim.x + threadIdx.x;
    if (i >= Bb * 2 * Hh) return;
    int b = i / (2 * Hh), c = i % (2 * Hh);
    emb[i] = (c < Hh) ? mean[b * Hh + c] : gmax[b * Hh + c - Hh];
}

__global__ void k_zcat(const float* __restrict__ emb, const float* __restrict__ gfeat,
                       float* __restrict__ z) {
    const int CD = 2 * Hh + GFf;
    int i = blockIdx.x * blockDim.x + threadIdx.x;
    if (i >= Bb * CD) return;
    int b = i / CD, c = i % CD;
    z[i] = (c < 2 * Hh) ? emb[b * 2 * Hh + c] : gfeat[b * GFf + c - 2 * Hh];
}

// ---------------- heads ----------------

__global__ void k_linear(const float* __restrict__ in, const float* __restrict__ W,
                         const float* __restrict__ bias, float* __restrict__ out,
                         int R, int K, int C, int relu) {
    int i = blockIdx.x * blockDim.x + threadIdx.x;
    if (i >= R * C) return;
    int r = i / C, c = i % C;
    float acc = bias[c];
    for (int k = 0; k < K; k++) acc += in[r * K + k] * W[k * C + c];
    out[i] = relu ? fmaxf(acc, 0.f) : acc;
}

__global__ void k_actor_out(const float* __restrict__ lin, float* __restrict__ logits,
                            float* __restrict__ probs) {
    int b = blockIdx.x * blockDim.x + threadIdx.x;
    if (b >= Bb) return;
    float v[NAa];
    float mx = -3.0e38f;
#pragma unroll
    for (int j = 0; j < NAa; j++) { v[j] = lin[b * NAa + j]; mx = fmaxf(mx, v[j]); }
    float s = 0.f;
#pragma unroll
    for (int j = 0; j < NAa; j++) { v[j] = expf(v[j] - mx); s += v[j]; }
#pragma unroll
    for (int j = 0; j < NAa; j++) {
        logits[b * NAa + j] = lin[b * NAa + j];
        probs[b * NAa + j] = v[j] / s;
    }
}

__global__ void k_critic_out(const float* __restrict__ co, float* __restrict__ value,
                             float* __restrict__ term) {
    int b = blockIdx.x * blockDim.x + threadIdx.x;
    if (b >= Bb) return;
    value[b] = co[b * 2];
    term[b] = 1.f / (1.f + expf(-co[b * 2 + 1]));
}

// ---------------- host orchestration ----------------

struct EncP {
    const float* convW[3];
    const float* convb[3];
    const float* inW;
    const float* inb;
    const float* na[3];
    const float* nb[3];
    const float* nw[3];
};
struct HeadP {
    const float* W1; const float* W2; const float* W3;
    const float* b1; const float* b2; const float* b3;
};

extern "C" void kernel_launch(void* const* d_in, const int* in_sizes, int n_in,
                              void* d_out, int out_size, void* d_ws, size_t ws_size,
                              hipStream_t stream) {
    (void)in_sizes; (void)n_in; (void)out_size; (void)ws_size;
    const int T = 256;
    const int NHB = (Nn * Hh + T - 1) / T;
    const int NB  = (Nn + T - 1) / T;
    const int EB  = (Ee + T - 1) / T;
    const int BHB = (Bb * Hh + T - 1) / T;
    constexpr int NC = (Nn + CHK - 1) / CHK;
    const int SEGB = (NC * Hh + T - 1) / T;

    auto F = [&](int i) { return (const float*)d_in[i]; };
    const float* x     = F(0);
    const int*   edge  = (const int*)d_in[1];
    const int*   batch = (const int*)d_in[2];
    const float* gfeat = F(3);

    // pytree (sorted-key) flattening: actor_enc @4, actor_head @21, critic_enc @27, critic_head @44
    auto loadEnc = [&](int base) {
        EncP p;
        for (int i = 0; i < 3; i++) {
            p.convW[i] = F(base + i);
            p.convb[i] = F(base + 3 + i);
            p.na[i]    = F(base + 8 + i);
            p.nb[i]    = F(base + 11 + i);
            p.nw[i]    = F(base + 14 + i);
        }
        p.inW = F(base + 6);
        p.inb = F(base + 7);
        return p;
    };
    auto loadHead = [&](int base) {
        HeadP h;
        h.W1 = F(base); h.W2 = F(base + 1); h.W3 = F(base + 2);
        h.b1 = F(base + 3); h.b2 = F(base + 4); h.b3 = F(base + 5);
        return h;
    };
    EncP  encs[2]  = { loadEnc(4),  loadEnc(27) };
    HeadP heads[2] = { loadHead(21), loadHead(44) };

    // d_out layout (flat, return order)
    float* out      = (float*)d_out;
    float* o_logits = out;                         // 64*7
    float* o_probs  = out + Bb * NAa;              // 64*7
    float* o_value  = o_probs + Bb * NAa;          // 64
    float* o_term   = o_value + Bb;                // 64
    float* o_aemb   = o_term + Bb;                 // 64*256
    float* o_cemb   = o_aemb + Bb * 2 * Hh;        // 64*256
    float* o_anodes = o_cemb + Bb * 2 * Hh;        // 50000*128
    float* o_cnodes = o_anodes + (long long)Nn * Hh;
    float* nodesOut[2] = { o_anodes, o_cnodes };
    float* embOut[2]   = { o_aemb, o_cemb };

    // workspace bump allocator (256B aligned)
    char* wsc = (char*)d_ws;
    size_t off = 0;
    auto alloc = [&](size_t bytes) -> void* {
        off = (off + 255) & ~(size_t)255;
        void* p = wsc + off;
        off += bytes;
        return p;
    };
    float* deg    = (float*)alloc((size_t)Nn * 4);
    float* dinv   = (float*)alloc((size_t)Nn * 4);
    unsigned short* hbf = (unsigned short*)alloc((size_t)Nn * Hh * 2);
    float* m      = (float*)alloc((size_t)Nn * Hh * 4);
    float* agg    = (float*)alloc((size_t)Nn * Hh * 4);
    unsigned short* Wt  = (unsigned short*)alloc((size_t)Hh * Hh * 2);
    float* segsum = (float*)alloc((size_t)Bb * Hh * 4);
    float* meanb  = (float*)alloc((size_t)Bb * Hh * 4);
    float* varsum = (float*)alloc((size_t)Bb * Hh * 4);
    float* rsb    = (float*)alloc((size_t)Bb * Hh * 4);
    float* gmaxb  = (float*)alloc((size_t)Bb * Hh * 4);
    float* cnt    = (float*)alloc((size_t)Bb * 4);
    float* zcat   = (float*)alloc((size_t)Bb * (2 * Hh + GFf) * 4);
    float* z1     = (float*)alloc((size_t)Bb * Hh * 4);
    float* z2     = (float*)alloc((size_t)Bb * (Hh / 2) * 4);
    float* coutb  = (float*)alloc((size_t)Bb * NAa * 4);

    // shared setup: degree normalization + segment counts (layer-invariant)
    hipMemsetAsync(deg, 0, (size_t)Nn * 4, stream);
    hipMemsetAsync(cnt, 0, (size_t)Bb * 4, stream);
    k_deg<<<EB, T, 0, stream>>>(edge + Ee, deg);
    k_dinv<<<NB, T, 0, stream>>>(deg, dinv);
    k_seg_cnt<<<NB, T, 0, stream>>>(batch, cnt);

    for (int enc = 0; enc < 2; enc++) {
        const EncP& p = encs[enc];
        float* h = nodesOut[enc];   // node features live in their d_out slot

        k_input_linear<<<NHB, T, 0, stream>>>(x, p.inW, p.inb, h, hbf);

        for (int l = 0; l < 3; l++) {
            k_wT_bf16<<<(Hh * Hh + T - 1) / T, T, 0, stream>>>(p.convW[l], Wt);
            // m = h @ W via bf16 WMMA; agg = m*dinv^2 + conv_b fused in epilogue
            k_gemm_bf16<<<(NTILES + 7) / 8, T, 0, stream>>>(hbf, Wt, dinv, p.convb[l], m, agg);
            k_edge_scatter<<<(Ee * 32 + T - 1) / T, T, 0, stream>>>(edge, m, dinv, agg);
            // GroupNorm over batch segments
            hipMemsetAsync(segsum, 0, (size_t)Bb * Hh * 4, stream);
            hipMemsetAsync(varsum, 0, (size_t)Bb * Hh * 4, stream);
            k_seg_sum<<<SEGB, T, 0, stream>>>(agg, batch, segsum);
            k_mean<<<BHB, T, 0, stream>>>(segsum, cnt, meanb);
            k_var_acc<<<SEGB, T, 0, stream>>>(agg, batch, meanb, p.na[l], varsum);
            k_rs<<<BHB, T, 0, stream>>>(varsum, cnt, rsb);
            k_norm_finish<<<NHB, T, 0, stream>>>(agg, batch, meanb, rsb,
                                                 p.na[l], p.nw[l], p.nb[l], h, hbf);
        }

        // pooling: mean + max per segment -> emb (in d_out)
        hipMemsetAsync(segsum, 0, (size_t)Bb * Hh * 4, stream);
        k_fill<<<BHB, T, 0, stream>>>(gmaxb, -3.0e38f, Bb * Hh);
        k_seg_sum<<<SEGB, T, 0, stream>>>(h, batch, segsum);
        k_mean<<<BHB, T, 0, stream>>>(segsum, cnt, meanb);
        k_gmax<<<SEGB, T, 0, stream>>>(h, batch, gmaxb);
        k_emb<<<(Bb * 2 * Hh + T - 1) / T, T, 0, stream>>>(meanb, gmaxb, embOut[enc]);

        // head
        const HeadP& hp = heads[enc];
        const int CD = 2 * Hh + GFf;
        k_zcat<<<(Bb * CD + T - 1) / T, T, 0, stream>>>(embOut[enc], gfeat, zcat);
        k_linear<<<(Bb * Hh + T - 1) / T, T, 0, stream>>>(zcat, hp.W1, hp.b1, z1, Bb, CD, Hh, 1);
        k_linear<<<(Bb * (Hh / 2) + T - 1) / T, T, 0, stream>>>(z1, hp.W2, hp.b2, z2, Bb, Hh, Hh / 2, 1);
        int outd = (enc == 0) ? NAa : 2;
        k_linear<<<(Bb * outd + T - 1) / T, T, 0, stream>>>(z2, hp.W3, hp.b3, coutb, Bb, Hh / 2, outd, 0);
        if (enc == 0) {
            k_actor_out<<<(Bb + T - 1) / T, T, 0, stream>>>(coutb, o_logits, o_probs);
        } else {
            k_critic_out<<<(Bb + T - 1) / T, T, 0, stream>>>(coutb, o_value, o_term);
        }
    }
}
